// DeepSpeedMLP_85487029059846
// MI455X (gfx1250) — matrix operand order repaired and checked by reading, not compiled, not run on hardware
//
#include <hip/hip_runtime.h>
#include <stdint.h>

typedef __attribute__((ext_vector_type(8)))  __bf16 v8bf;
typedef __attribute__((ext_vector_type(16))) __bf16 v16bf;
typedef __attribute__((ext_vector_type(8)))  float  v8f;
typedef __attribute__((ext_vector_type(2)))  __bf16 v2bf;
typedef __attribute__((ext_vector_type(4)))  int    v4i;

#if __has_builtin(__builtin_amdgcn_cvt_pk_bf16_f32)
#define HAVE_PK_BF16 1
#else
#define HAVE_PK_BF16 0
#endif

#if __has_builtin(__builtin_amdgcn_global_load_async_to_lds_b128) && \
    __has_builtin(__builtin_amdgcn_s_wait_asynccnt)
#define HAVE_ASYNC_LDS 1
#else
#define HAVE_ASYNC_LDS 0
#endif

// scalar f32 -> bf16 (RNE), kept in 32-bit regs
__device__ __forceinline__ uint16_t f2bf(float f) {
    uint32_t u = __float_as_uint(f);
    u += 0x7FFFu + ((u >> 16) & 1u);
    return (uint16_t)(u >> 16);
}

// pack two f32 -> packed bf16 pair (lo in [15:0], hi in [31:16]); RNE
__device__ __forceinline__ uint32_t pack2bf(float lo, float hi) {
#if HAVE_PK_BF16
    v2bf p = __builtin_amdgcn_cvt_pk_bf16_f32(lo, hi);
    return __builtin_bit_cast(uint32_t, p);
#else
    uint32_t ul = __float_as_uint(lo);
    uint32_t uh = __float_as_uint(hi);
    ul += 0x7FFFu + ((ul >> 16) & 1u);
    uh += 0x7FFFu + ((uh >> 16) & 1u);
#if __has_builtin(__builtin_amdgcn_perm)
    return __builtin_amdgcn_perm(uh, ul, 0x07060302u);  // {uh[3:2], ul[3:2]}
#else
    return (ul >> 16) | (uh & 0xFFFF0000u);
#endif
#endif
}

#define LN_H 4096

// ---------------------------------------------------------------------------
// Kernel 1: fused residual-add + LayerNorm, bf16 output (one block per token)
// ---------------------------------------------------------------------------
__global__ __launch_bounds__(256) void fused_add_ln_bf16(
    const float* __restrict__ x, const float* __restrict__ res,
    const float* __restrict__ gamma, const float* __restrict__ beta,
    uint16_t* __restrict__ lnout)
{
    const int row = blockIdx.x;
    const int tid = threadIdx.x;
    const float4* xr = (const float4*)(x   + (size_t)row * LN_H);
    const float4* rr = (const float4*)(res + (size_t)row * LN_H);

    float4 h[4];
    float s = 0.f, s2 = 0.f;
#pragma unroll
    for (int i = 0; i < 4; ++i) {
        int idx = tid + i * 256;
        float4 a = xr[idx], b = rr[idx];
        float4 t = make_float4(a.x + b.x, a.y + b.y, a.z + b.z, a.w + b.w);
        h[i] = t;
        s  += t.x + t.y + t.z + t.w;
        s2 += t.x * t.x + t.y * t.y + t.z * t.z + t.w * t.w;
    }
#pragma unroll
    for (int off = 16; off > 0; off >>= 1) {
        s  += __shfl_down(s,  off, 32);
        s2 += __shfl_down(s2, off, 32);
    }
    __shared__ float redS[8], redS2[8], mv[2];
    const int lane = tid & 31, wvid = tid >> 5;
    if (lane == 0) { redS[wvid] = s; redS2[wvid] = s2; }
    __syncthreads();
    if (tid == 0) {
        float ts = 0.f, ts2 = 0.f;
        for (int w = 0; w < 8; ++w) { ts += redS[w]; ts2 += redS2[w]; }
        float mu  = ts * (1.f / LN_H);
        float var = ts2 * (1.f / LN_H) - mu * mu;
        mv[0] = mu;
        mv[1] = rsqrtf(var + 1e-5f);
    }
    __syncthreads();
    const float mu = mv[0], rstd = mv[1];

    const float4* g4 = (const float4*)gamma;
    const float4* b4 = (const float4*)beta;
    uint2* outp = (uint2*)(lnout + (size_t)row * LN_H);
#pragma unroll
    for (int i = 0; i < 4; ++i) {
        int idx = tid + i * 256;
        float4 g = g4[idx], b = b4[idx];
        float4 t = h[i];
        float o0 = (t.x - mu) * rstd * g.x + b.x;
        float o1 = (t.y - mu) * rstd * g.y + b.y;
        float o2 = (t.z - mu) * rstd * g.z + b.z;
        float o3 = (t.w - mu) * rstd * g.w + b.w;
        uint2 p;
        p.x = pack2bf(o0, o1);
        p.y = pack2bf(o2, o3);
        outp[idx] = p;
    }
}

// ---------------------------------------------------------------------------
// Kernel 2/3: bf16 WMMA GEMM  C[M,N] = A[M,K](bf16) * W[K,N](f32->bf16)
// Block tile 128x256, BK=32, double-buffered LDS.
// 8 waves (2x4); each wave: 64x64 output = 4x4 wmma tiles (128 acc VGPRs).
// EPI==0: +bias, ReLU, store bf16.   EPI==1: +bias +x +residual, store f32.
// ---------------------------------------------------------------------------
#define BM 128
#define BN 256
#define BK 32
#define LDSS 40   // padded row stride in bf16 elems (kills ds bank conflicts)

template <int EPI>
__global__ __launch_bounds__(256) void gemm_bf16_wmma(
    const uint16_t* __restrict__ A,   // [M,K] bf16 row-major
    const float*    __restrict__ W,   // [K,N] f32 row-major
    const float*    __restrict__ bias,// [N]
    const float*    __restrict__ addx,// EPI==1: x [M,N]
    const float*    __restrict__ addr,// EPI==1: residual [M,N]
    void*           __restrict__ outp,
    int M, int N, int K)
{
    __shared__ uint16_t ldsA[2][BM * LDSS];   // [row][k]
    __shared__ uint16_t ldsB[2][BN * LDSS];   // [n][k] (transposed on stage)

    const int tid   = threadIdx.x;
    const int lane  = tid & 31;
    const int wvid  = tid >> 5;
    const int waveM = wvid >> 2;               // 0..1  -> 64-row strip
    const int waveN = wvid & 3;                // 0..3  -> 64-col strip
    const int tileM = blockIdx.x * BM;         // M fast axis: weight reuse in L2
    const int tileN = blockIdx.y * BN;
    const int  lmod = lane & 15;
    const bool lhi  = lane >= 16;

    v8f acc[4][4];
#pragma unroll
    for (int i = 0; i < 4; ++i)
#pragma unroll
        for (int j = 0; j < 4; ++j) acc[i][j] = {};

    // A staging: 2 chunks of 16B per thread (rows ar0 and ar0+64)
    const int ar0 = tid >> 2;
    const int ac0 = (tid & 3) * 8;
    // B staging: thread -> (k pair, 16 columns)
    const int kp = tid >> 4;     // k0 = kp*2
    const int ng = tid & 15;     // n0 = ng*16

    const int kSteps = K / BK;

    float wr0[16], wr1[16];              // W prefetch regs (two k rows x 16 n)
#if !HAVE_ASYNC_LDS
    uint4 aR0, aR1;                      // A prefetch regs (fallback path)
#endif

    auto globalFetch = [&](int ks) {
        const int k0 = ks * BK;
#if !HAVE_ASYNC_LDS
        aR0 = *(const uint4*)(A + (size_t)(tileM + ar0) * K + k0 + ac0);
        aR1 = *(const uint4*)(A + (size_t)(tileM + ar0 + 64) * K + k0 + ac0);
#endif
        const float* w0 = W + (size_t)(k0 + kp * 2) * N + tileN + ng * 16;
        const float* w1 = w0 + N;
#pragma unroll
        for (int q = 0; q < 4; ++q) {
            float4 t0 = ((const float4*)w0)[q];
            float4 t1 = ((const float4*)w1)[q];
            wr0[q * 4 + 0] = t0.x; wr0[q * 4 + 1] = t0.y;
            wr0[q * 4 + 2] = t0.z; wr0[q * 4 + 3] = t0.w;
            wr1[q * 4 + 0] = t1.x; wr1[q * 4 + 1] = t1.y;
            wr1[q * 4 + 2] = t1.z; wr1[q * 4 + 3] = t1.w;
        }
    };

#if HAVE_ASYNC_LDS
    auto asyncA = [&](int ks, int buf) {
        const int k0 = ks * BK;
        __builtin_amdgcn_global_load_async_to_lds_b128(
            (__attribute__((address_space(1))) v4i*)(A + (size_t)(tileM + ar0) * K + k0 + ac0),
            (__attribute__((address_space(3))) v4i*)&ldsA[buf][ar0 * LDSS + ac0],
            0, 0);
        __builtin_amdgcn_global_load_async_to_lds_b128(
            (__attribute__((address_space(1))) v4i*)(A + (size_t)(tileM + ar0 + 64) * K + k0 + ac0),
            (__attribute__((address_space(3))) v4i*)&ldsA[buf][(ar0 + 64) * LDSS + ac0],
            0, 0);
    };
#endif

    auto ldsStore = [&](int buf) {
#if !HAVE_ASYNC_LDS
        *(uint4*)&ldsA[buf][ar0 * LDSS + ac0]        = aR0;
        *(uint4*)&ldsA[buf][(ar0 + 64) * LDSS + ac0] = aR1;
#endif
#pragma unroll
        for (int j = 0; j < 16; ++j)
            *(uint32_t*)&ldsB[buf][(ng * 16 + j) * LDSS + kp * 2] = pack2bf(wr0[j], wr1[j]);
    };

    auto compute = [&](int buf) {
        v16bf af[4], bfr[4];
#pragma unroll
        for (int ms = 0; ms < 4; ++ms) {
            int ar  = waveM * 64 + ms * 16 + lmod;
            int akb = lhi ? 8 : 0;                    // lanes 16-31: K=8..15 / 24..31
            v8bf lo = *(const v8bf*)&ldsA[buf][ar * LDSS + akb];
            v8bf hi = *(const v8bf*)&ldsA[buf][ar * LDSS + akb + 16];
            af[ms] = __builtin_shufflevector(lo, hi,
                0,1,2,3,4,5,6,7,8,9,10,11,12,13,14,15);
        }
#pragma unroll
        for (int ns = 0; ns < 4; ++ns) {
            int bn  = waveN * 64 + ns * 16 + lmod;
            int bkb = lhi ? 8 : 0;                    // lanes 16-31: K=8..15 / 24..31
            v8bf lo = *(const v8bf*)&ldsB[buf][bn * LDSS + bkb];
            v8bf hi = *(const v8bf*)&ldsB[buf][bn * LDSS + bkb + 16];
            bfr[ns] = __builtin_shufflevector(lo, hi,
                0,1,2,3,4,5,6,7,8,9,10,11,12,13,14,15);
        }
#pragma unroll
        for (int ms = 0; ms < 4; ++ms)
#pragma unroll
            for (int ns = 0; ns < 4; ++ns)
                acc[ms][ns] = __builtin_amdgcn_wmma_f32_16x16x32_bf16(
                    false, af[ms], false, bfr[ns],
                    (short)0, acc[ms][ns], false, false);
    };

    // ---- pipeline prologue: stage tile 0 ----
#if HAVE_ASYNC_LDS
    asyncA(0, 0);
#endif
    globalFetch(0);
    ldsStore(0);
#if HAVE_ASYNC_LDS
    __builtin_amdgcn_s_wait_asynccnt(0);
#endif
    __syncthreads();

    // ---- main loop: one barrier per k-step, global loads overlap wmma ----
    for (int ks = 0; ks < kSteps; ++ks) {
        const int cur  = ks & 1;
        const bool more = (ks + 1) < kSteps;
        if (more) {
#if HAVE_ASYNC_LDS
            asyncA(ks + 1, cur ^ 1);          // lds[nxt] free since barrier ks-1
#endif
            globalFetch(ks + 1);
        }
        compute(cur);
        if (more) {
            ldsStore(cur ^ 1);
#if HAVE_ASYNC_LDS
            __builtin_amdgcn_s_wait_asynccnt(0);
#endif
        }
        __syncthreads();
    }

    // ---- epilogue (C/D layout: vgpr r -> M = r + (lane<16?0:8), N = lane&15) ----
    const int rowBase = tileM + waveM * 64 + (lhi ? 8 : 0);
#pragma unroll
    for (int ns = 0; ns < 4; ++ns) {
        const int col = tileN + waveN * 64 + ns * 16 + lmod;
        const float bv = bias[col];
#pragma unroll
        for (int ms = 0; ms < 4; ++ms) {
#pragma unroll
            for (int r = 0; r < 8; ++r) {
                const int row = rowBase + ms * 16 + r;
                const size_t off = (size_t)row * N + col;
                float v = acc[ms][ns][r] + bv;
                if (EPI == 0) {
                    v = v > 0.f ? v : 0.f;                 // ReLU
                    ((uint16_t*)outp)[off] = f2bf(v);      // bf16 activation
                } else {
                    v += addx[off] + addr[off];            // + x + residual
                    ((float*)outp)[off] = v;
                }
            }
        }
    }
}

// ---------------------------------------------------------------------------
extern "C" void kernel_launch(void* const* d_in, const int* in_sizes, int n_in,
                              void* d_out, int out_size, void* d_ws, size_t ws_size,
                              hipStream_t stream) {
    const float* x      = (const float*)d_in[0];
    const float* resid  = (const float*)d_in[1];
    const float* gamma  = (const float*)d_in[2];
    const float* beta   = (const float*)d_in[3];
    const float* interw = (const float*)d_in[4];
    const float* interb = (const float*)d_in[5];
    const float* outw   = (const float*)d_in[6];
    const float* outb   = (const float*)d_in[7];

    const int M = 4096, H = 4096, I = 16384;

    uint16_t* lnbuf = (uint16_t*)d_ws;                  // [M,H] bf16 (32 MB)
    uint16_t* act   = lnbuf + (size_t)M * H;            // [M,I] bf16 (128 MB)

    fused_add_ln_bf16<<<M, 256, 0, stream>>>(x, resid, gamma, beta, lnbuf);

    // fc1 + ReLU -> bf16 activations.  grid.x = M (fast) for L2 weight reuse.
    gemm_bf16_wmma<0><<<dim3(M / BM, I / BN), 256, 0, stream>>>(
        lnbuf, interw, interb, nullptr, nullptr, act, M, I, H);

    // fc2 + bias + x + residual -> f32 output
    gemm_bf16_wmma<1><<<dim3(M / BM, H / BN), 256, 0, stream>>>(
        act, outw, outb, x, resid, d_out, M, H, I);
}
